// SwinTransformerBlock_55937654063208
// MI455X (gfx1250) — matrix operand order repaired and checked by reading, not compiled, not run on hardware
//
#include <hip/hip_runtime.h>
#include <hip/hip_bf16.h>

// ---------------------------------------------------------------------------
// Swin Transformer block on gfx1250 (MI455X), WMMA f16 path.
// B=16, H=W=56, C=384, heads=12, window 7x7 (N=49 padded to 64), MLP=1536.
// Round 3: fix async-LDS builtin pointer types (v4i addrspace(1)/(3) pointers,
// per the clang diagnostic from round 2).
// ---------------------------------------------------------------------------

typedef __attribute__((ext_vector_type(16))) _Float16 v16h;
typedef __attribute__((ext_vector_type(8)))  _Float16 v8h;
typedef __attribute__((ext_vector_type(8)))  float    v8f;
typedef __attribute__((ext_vector_type(4)))  int      v4i;

union U16h { v16h v; v8h h[2]; };

#if defined(__has_builtin)
#if __has_builtin(__builtin_amdgcn_global_load_async_to_lds_b128)
#define USE_ASYNC_LDS 1
#endif
#endif
#ifndef USE_ASYNC_LDS
#define USE_ASYNC_LDS 0
#endif

static __device__ inline void wait_async0() {
#if defined(__has_builtin) && __has_builtin(__builtin_amdgcn_s_wait_asynccnt)
  __builtin_amdgcn_s_wait_asynccnt(0);
#else
  asm volatile("s_wait_asynccnt 0x0" ::: "memory");
#endif
}

#if USE_ASYNC_LDS
typedef __attribute__((address_space(1))) v4i gbl_v4i;
typedef __attribute__((address_space(3))) v4i lds_v4i;
// memory -> LDS DMA, 16 bytes per lane, tracked with ASYNCcnt (no VGPR data).
static __device__ inline void async_copy16(const _Float16* g, _Float16* l) {
  __builtin_amdgcn_global_load_async_to_lds_b128(
      (gbl_v4i*)g, (lds_v4i*)l, /*imm offset=*/0, /*cpol=*/0);
}
#endif

static __device__ inline v8f v8f_zero() {
  v8f z = {0.f, 0.f, 0.f, 0.f, 0.f, 0.f, 0.f, 0.f};
  return z;
}

static __device__ inline v16h ld_pair(const _Float16* p0, const _Float16* p1) {
  U16h u;
  u.h[0] = *(const v8h*)p0;
  u.h[1] = *(const v8h*)p1;
  return u.v;
}

// A operand (16x32 f16) from row-major [M][K] source, leading dim ld (halves).
// ISA layout: lane l (m = l&15, half = l>>4); elems 0..7 -> K = half*8 + 0..7,
// elems 8..15 -> K = 16 + half*8 + 0..7.
static __device__ inline v16h load_A(const _Float16* base, int ld, int row0, int k0) {
  const int l = threadIdx.x & 31;
  const int m = l & 15;
  const int hf = l >> 4;
  const _Float16* p = base + (size_t)(row0 + m) * ld + k0 + hf * 8;
  return ld_pair(p, p + 16);
}

// B operand (32x16 f16) from B^T row-major [N][K], leading dim ld (halves).
// ISA layout: lane l (n = l&15, half = l>>4); elems 0..7 -> K = half*8 + 0..7,
// elems 8..15 -> K = 16 + half*8 + 0..7.
static __device__ inline v16h load_Bt(const _Float16* base, int ld, int col0, int k0) {
  const int l = threadIdx.x & 31;
  const int n = l & 15;
  const int hf = l >> 4;
  const _Float16* p = base + (size_t)(col0 + n) * ld + k0 + hf * 8;
  return ld_pair(p, p + 16);
}

static __device__ inline v8f wmma_f16(v16h a, v16h b, v8f c) {
  return __builtin_amdgcn_wmma_f32_16x16x32_f16(
      /*neg_a=*/false, a, /*neg_b=*/false, b,
      /*c_mod=*/(short)0, c, /*reuse_a=*/false, /*reuse_b=*/false);
}

// ---------------------------------------------------------------------------
// 64x128 macro-tile GEMM core. 256 threads = 8 waves in a 4x2 grid.
// sA: 2 buffers of 64x32 halves (8KB), sB: 2 buffers of 128x32 halves (16KB).
// Double-buffered K-pipeline: stage tile t+1 while computing tile t.
// ---------------------------------------------------------------------------
static __device__ inline void gemm_tile_64x128(const _Float16* __restrict__ A, int lda,
                                               const _Float16* __restrict__ Bt, int ldb,
                                               int K, _Float16* sA, _Float16* sB,
                                               v8f acc[4]) {
  const int tid = threadIdx.x;
  const int w = tid >> 5;   // wave 0..7
  const int mr = w >> 1;    // row tile 0..3
  const int nc = w & 1;     // col half 0..1
#pragma unroll
  for (int i = 0; i < 4; ++i) acc[i] = v8f_zero();

  const int a_row = tid >> 2, a_kc = (tid & 3) * 8;   // A stage: 8 halves/thread
  const int b_col = tid >> 1, b_kc = (tid & 1) * 16;  // B stage: 16 halves/thread
  const _Float16* ga = A + (size_t)a_row * lda + a_kc;
  const _Float16* gb = Bt + (size_t)b_col * ldb + b_kc;
  const int sa_off = a_row * 32 + a_kc;
  const int sb_off = b_col * 32 + b_kc;

#if USE_ASYNC_LDS
  // prologue: DMA tile 0 into buffer 0 (no VGPR round-trip, ASYNCcnt tracked)
  async_copy16(ga, sA + sa_off);
  async_copy16(gb, sB + sb_off);
  async_copy16(gb + 8, sB + sb_off + 8);
  wait_async0();
  __syncthreads();
  for (int k0 = 0; k0 < K; k0 += 32) {
    const int cur = (k0 >> 5) & 1;
    const _Float16* cA = sA + cur * (64 * 32);
    const _Float16* cB = sB + cur * (128 * 32);
    const bool more = (k0 + 32) < K;
    if (more) {  // kick DMA for next tile into the other buffer
      const int nxt = cur ^ 1;
      async_copy16(ga + k0 + 32, sA + nxt * (64 * 32) + sa_off);
      async_copy16(gb + k0 + 32, sB + nxt * (128 * 32) + sb_off);
      async_copy16(gb + k0 + 40, sB + nxt * (128 * 32) + sb_off + 8);
    }
    v16h av = load_A(cA, 32, mr * 16, 0);
#pragma unroll
    for (int nt = 0; nt < 4; ++nt) {
      v16h bv = load_Bt(cB, 32, nc * 64 + nt * 16, 0);
      acc[nt] = wmma_f16(av, bv, acc[nt]);
    }
    if (more) {
      wait_async0();
      __syncthreads();
    }
  }
#else
  // fallback: register-pipelined double buffer
  v8h ra = *(const v8h*)ga;
  v8h rb0 = *(const v8h*)gb;
  v8h rb1 = *(const v8h*)(gb + 8);
  *(v8h*)(sA + sa_off) = ra;
  *(v8h*)(sB + sb_off) = rb0;
  *(v8h*)(sB + sb_off + 8) = rb1;
  __syncthreads();
  for (int k0 = 0; k0 < K; k0 += 32) {
    const int cur = (k0 >> 5) & 1;
    const _Float16* cA = sA + cur * (64 * 32);
    const _Float16* cB = sB + cur * (128 * 32);
    const bool more = (k0 + 32) < K;
    v8h na = ra, nb0 = rb0, nb1 = rb1;
    if (more) {  // issue next-tile global loads early
      na = *(const v8h*)(ga + k0 + 32);
      nb0 = *(const v8h*)(gb + k0 + 32);
      nb1 = *(const v8h*)(gb + k0 + 40);
      __builtin_prefetch(ga + k0 + 64, 0, 1);
      __builtin_prefetch(gb + k0 + 64, 0, 1);
    }
    v16h av = load_A(cA, 32, mr * 16, 0);
#pragma unroll
    for (int nt = 0; nt < 4; ++nt) {
      v16h bv = load_Bt(cB, 32, nc * 64 + nt * 16, 0);
      acc[nt] = wmma_f16(av, bv, acc[nt]);
    }
    if (more) {
      const int nxt = cur ^ 1;
      *(v8h*)(sA + nxt * (64 * 32) + sa_off) = na;
      *(v8h*)(sB + nxt * (128 * 32) + sb_off) = nb0;
      *(v8h*)(sB + nxt * (128 * 32) + sb_off + 8) = nb1;
      __syncthreads();
    }
  }
#endif
}

// ---------------------------------------------------------------------------
// Weight convert + transpose: dst[n*K + k] = (f16) src[k*N + n]
// ---------------------------------------------------------------------------
__global__ __launch_bounds__(256) void conv_transpose_kernel(const float* __restrict__ src,
                                                             _Float16* __restrict__ dst,
                                                             int K, int N) {
  int i = blockIdx.x * 256 + threadIdx.x;
  if (i < K * N) {
    int k = i / N, n = i - k * N;
    dst[(size_t)n * K + k] = (_Float16)src[i];
  }
}

// ---------------------------------------------------------------------------
// LayerNorm stats over H*W*C per batch sample. grid = B, block = 256.
// ---------------------------------------------------------------------------
__global__ __launch_bounds__(256) void ln_stats_kernel(const float* __restrict__ x,
                                                       float* __restrict__ stats) {
  const int b = blockIdx.x;
  const size_t n = (size_t)56 * 56 * 384;
  const float* p = x + (size_t)b * n;
  float s = 0.f, ss = 0.f;
  for (size_t i = threadIdx.x; i < n; i += 256) {
    float v = p[i];
    s += v;
    ss += v * v;
  }
  __shared__ float sh1[256], sh2[256];
  sh1[threadIdx.x] = s;
  sh2[threadIdx.x] = ss;
  __syncthreads();
  for (int o = 128; o > 0; o >>= 1) {
    if (threadIdx.x < o) {
      sh1[threadIdx.x] += sh1[threadIdx.x + o];
      sh2[threadIdx.x] += sh2[threadIdx.x + o];
    }
    __syncthreads();
  }
  if (threadIdx.x == 0) {
    float inv_n = 1.f / (float)n;
    float mean = sh1[0] * inv_n;
    float var = sh2[0] * inv_n - mean * mean;
    stats[b * 2 + 0] = mean;
    stats[b * 2 + 1] = rsqrtf(var + 1e-5f);
  }
}

// ---------------------------------------------------------------------------
// LN1 apply + shift-roll(-3,-3) + window partition + f16 + pad rows to 64.
// grid = 1024 windows, block = 256. xw: [win][64][384] f16.
// ---------------------------------------------------------------------------
__global__ __launch_bounds__(256) void ln1_partition_kernel(const float* __restrict__ x,
                                                            const float* __restrict__ stats,
                                                            const float* __restrict__ sc,
                                                            const float* __restrict__ bi,
                                                            _Float16* __restrict__ xw) {
  const int win = blockIdx.x;
  const int b = win >> 6, wr = (win >> 3) & 7, wc = win & 7;
  const float mean = stats[b * 2], rstd = stats[b * 2 + 1];
  _Float16* dst = xw + (size_t)win * 64 * 384;
  for (int idx = threadIdx.x; idx < 64 * 384; idx += 256) {
    int row = idx / 384, c = idx - row * 384;
    _Float16 o = (_Float16)0.f;
    if (row < 49) {
      int i = row / 7, j = row - i * 7;
      int hs = (wr * 7 + i + 3) % 56;  // shifted[h] = x[(h+3)%56]
      int ws2 = (wc * 7 + j + 3) % 56;
      float v = x[(((size_t)b * 56 + hs) * 56 + ws2) * 384 + c];
      o = (_Float16)((v - mean) * rstd * sc[c] + bi[c]);
    }
    dst[idx] = o;
  }
}

// ---------------------------------------------------------------------------
// QKV GEMM: per window M=64, N=1152, K=384.  grid (9, 1024), block 256.
// Scatters into q/k [win][head][64][32] and v^T [win][head][32][64].
// ---------------------------------------------------------------------------
__global__ __launch_bounds__(256) void qkv_gemm_kernel(const _Float16* __restrict__ xw,
                                                       const _Float16* __restrict__ wt,
                                                       const float* __restrict__ qkv_b,
                                                       _Float16* __restrict__ q,
                                                       _Float16* __restrict__ k,
                                                       _Float16* __restrict__ vt) {
  __shared__ __align__(16) _Float16 sA[2 * 64 * 32];
  __shared__ __align__(16) _Float16 sB[2 * 128 * 32];
  const int win = blockIdx.y;
  const int nblk = blockIdx.x;
  v8f acc[4];
  gemm_tile_64x128(xw + (size_t)win * 64 * 384, 384,
                   wt + (size_t)nblk * 128 * 384, 384, 384, sA, sB, acc);
  const int w = threadIdx.x >> 5, mr = w >> 1, nc = w & 1;
  const int l = threadIdx.x & 31, m16 = l & 15, hf = l >> 4;
#pragma unroll
  for (int nt = 0; nt < 4; ++nt) {
#pragma unroll
    for (int r = 0; r < 8; ++r) {
      int row = mr * 16 + hf * 8 + r;
      int n = nblk * 128 + nc * 64 + nt * 16 + m16;
      float v = acc[nt][r] + qkv_b[n];
      int which = n / 384;
      int rem = n - which * 384;
      int head = rem >> 5, d = rem & 31;
      size_t wh = (size_t)win * 12 + head;
      if (which == 0) {
        v *= 0.17677669529663687f;  // (hd=32)^-0.5
        q[(wh * 64 + row) * 32 + d] = (_Float16)v;
      } else if (which == 1) {
        k[(wh * 64 + row) * 32 + d] = (_Float16)v;
      } else {
        vt[(wh * 32 + d) * 64 + row] = (_Float16)v;
      }
    }
  }
}

// ---------------------------------------------------------------------------
// Fused window attention: one block (4 waves) per (window, head).
// scores = q k^T (WMMA) + rel_bias + mask -> softmax -> f16 via LDS -> A.V (WMMA)
// ---------------------------------------------------------------------------
__global__ __launch_bounds__(128) void attn_kernel(const _Float16* __restrict__ q,
                                                   const _Float16* __restrict__ k,
                                                   const _Float16* __restrict__ vt,
                                                   const float* __restrict__ bias_tab,
                                                   const float* __restrict__ maskp,
                                                   _Float16* __restrict__ out) {
  __shared__ __align__(16) _Float16 sAttn[64 * 64];
  const int wh = blockIdx.x;
  const int win = wh / 12, head = wh - win * 12;
  const int w = threadIdx.x >> 5;  // wave -> row tile 0..3
  const int l = threadIdx.x & 31, m16 = l & 15, hf = l >> 4;
  const _Float16* qb = q + (size_t)wh * 64 * 32;
  const _Float16* kb = k + (size_t)wh * 64 * 32;
  const _Float16* vb = vt + (size_t)wh * 32 * 64;

  // scores 16x64 per wave: K = 32 = head_dim, one WMMA per N-tile
  v16h av = load_A(qb, 32, w * 16, 0);
  v8f acc[4];
#pragma unroll
  for (int nt = 0; nt < 4; ++nt) {
    v16h bv = load_Bt(kb, 32, nt * 16, 0);
    acc[nt] = wmma_f16(av, bv, v8f_zero());
  }

  // bias + mask on C-layout elements (row = w*16 + hf*8 + r, col = nt*16 + m16)
  float vals[4][8];
#pragma unroll
  for (int nt = 0; nt < 4; ++nt) {
    int c = nt * 16 + m16;
#pragma unroll
    for (int r = 0; r < 8; ++r) {
      int row = w * 16 + hf * 8 + r;
      float x = acc[nt][r];
      if (row < 49 && c < 49) {
        int r0 = row / 7, c0 = row - r0 * 7;
        int r1 = c / 7, c1 = c - r1 * 7;
        int idx = (r0 - r1 + 6) * 13 + (c0 - c1 + 6);
        x += bias_tab[idx * 12 + head];
        x += maskp[(size_t)win * 2401 + row * 49 + c];
      } else {
        x = -1e30f;
      }
      vals[nt][r] = x;
    }
  }

  // softmax per row: local 4-col reduce, then xor-shuffle across the 16-lane group
#pragma unroll
  for (int r = 0; r < 8; ++r) {
    float mx = fmaxf(fmaxf(vals[0][r], vals[1][r]), fmaxf(vals[2][r], vals[3][r]));
    mx = fmaxf(mx, __shfl_xor(mx, 1, 32));
    mx = fmaxf(mx, __shfl_xor(mx, 2, 32));
    mx = fmaxf(mx, __shfl_xor(mx, 4, 32));
    mx = fmaxf(mx, __shfl_xor(mx, 8, 32));
    float e[4], s = 0.f;
#pragma unroll
    for (int nt = 0; nt < 4; ++nt) {
      e[nt] = __expf(vals[nt][r] - mx);
      s += e[nt];
    }
    s += __shfl_xor(s, 1, 32);
    s += __shfl_xor(s, 2, 32);
    s += __shfl_xor(s, 4, 32);
    s += __shfl_xor(s, 8, 32);
    float inv = (s > 0.f) ? (1.f / s) : 0.f;
    int row = w * 16 + hf * 8 + r;
#pragma unroll
    for (int nt = 0; nt < 4; ++nt) {
      sAttn[row * 64 + nt * 16 + m16] = (_Float16)(e[nt] * inv);
    }
  }
  __syncthreads();

  // out = attn @ v : M=16/wave, N=32, K=64 (2 k-steps of 32)
  v8f o[2] = {v8f_zero(), v8f_zero()};
#pragma unroll
  for (int kk = 0; kk < 2; ++kk) {
    v16h aa = load_A(sAttn, 64, w * 16, kk * 32);
#pragma unroll
    for (int nt = 0; nt < 2; ++nt) {
      v16h bb = load_Bt(vb, 64, nt * 16, kk * 32);
      o[nt] = wmma_f16(aa, bb, o[nt]);
    }
  }
  // write [win][row][head*32 + d] f16
#pragma unroll
  for (int nt = 0; nt < 2; ++nt) {
#pragma unroll
    for (int r = 0; r < 8; ++r) {
      int row = w * 16 + hf * 8 + r;
      int d = nt * 16 + m16;
      out[((size_t)win * 64 + row) * 384 + head * 32 + d] = (_Float16)o[nt][r];
    }
  }
}

// ---------------------------------------------------------------------------
// Proj GEMM + window reverse + unshift + residual. grid (3, 1024), block 256.
// ---------------------------------------------------------------------------
__global__ __launch_bounds__(256) void proj_gemm_kernel(const _Float16* __restrict__ ao,
                                                        const _Float16* __restrict__ wt,
                                                        const float* __restrict__ pb,
                                                        const float* __restrict__ x_in,
                                                        float* __restrict__ x1) {
  __shared__ __align__(16) _Float16 sA[2 * 64 * 32];
  __shared__ __align__(16) _Float16 sB[2 * 128 * 32];
  const int win = blockIdx.y;
  const int nblk = blockIdx.x;
  v8f acc[4];
  gemm_tile_64x128(ao + (size_t)win * 64 * 384, 384,
                   wt + (size_t)nblk * 128 * 384, 384, 384, sA, sB, acc);
  const int w = threadIdx.x >> 5, mr = w >> 1, nc = w & 1;
  const int l = threadIdx.x & 31, m16 = l & 15, hf = l >> 4;
  const int b = win >> 6, wr = (win >> 3) & 7, wc = win & 7;
#pragma unroll
  for (int nt = 0; nt < 4; ++nt) {
#pragma unroll
    for (int r = 0; r < 8; ++r) {
      int row = mr * 16 + hf * 8 + r;
      if (row < 49) {
        int c = nblk * 128 + nc * 64 + nt * 16 + m16;
        int i = row / 7, j = row - i * 7;
        int hh = (wr * 7 + i + 3) % 56;  // reverse roll(+3)
        int ww = (wc * 7 + j + 3) % 56;
        size_t off = (((size_t)b * 56 + hh) * 56 + ww) * 384 + c;
        x1[off] = x_in[off] + acc[nt][r] + pb[c];
      }
    }
  }
}

// ---------------------------------------------------------------------------
// LN2 apply (elementwise, uses stats2). one elem/thread, grid = 75264.
// ---------------------------------------------------------------------------
__global__ __launch_bounds__(256) void ln2_apply_kernel(const float* __restrict__ x1,
                                                        const float* __restrict__ stats,
                                                        const float* __restrict__ sc,
                                                        const float* __restrict__ bi,
                                                        _Float16* __restrict__ h2) {
  size_t i = (size_t)blockIdx.x * 256 + threadIdx.x;
  int b = (int)(i / ((size_t)3136 * 384));
  int c = (int)(i % 384);
  float mean = stats[b * 2], rstd = stats[b * 2 + 1];
  h2[i] = (_Float16)((x1[i] - mean) * rstd * sc[c] + bi[c]);
}

static __device__ inline float gelu_tanh(float x) {
  float x3 = x * x * x;
  return 0.5f * x * (1.f + tanhf(0.7978845608028654f * (x + 0.044715f * x3)));
}

// ---------------------------------------------------------------------------
// MLP1: M=50176, N=1536, K=384, bias + GELU -> f16. grid (12, 784).
// ---------------------------------------------------------------------------
__global__ __launch_bounds__(256) void mlp1_gemm_kernel(const _Float16* __restrict__ h2,
                                                        const _Float16* __restrict__ w1t,
                                                        const float* __restrict__ b1,
                                                        _Float16* __restrict__ hid) {
  __shared__ __align__(16) _Float16 sA[2 * 64 * 32];
  __shared__ __align__(16) _Float16 sB[2 * 128 * 32];
  const int mb = blockIdx.y;
  const int nblk = blockIdx.x;
  v8f acc[4];
  gemm_tile_64x128(h2 + (size_t)mb * 64 * 384, 384,
                   w1t + (size_t)nblk * 128 * 384, 384, 384, sA, sB, acc);
  const int w = threadIdx.x >> 5, mr = w >> 1, nc = w & 1;
  const int l = threadIdx.x & 31, m16 = l & 15, hf = l >> 4;
#pragma unroll
  for (int nt = 0; nt < 4; ++nt) {
#pragma unroll
    for (int r = 0; r < 8; ++r) {
      size_t row = (size_t)mb * 64 + mr * 16 + hf * 8 + r;
      int col = nblk * 128 + nc * 64 + nt * 16 + m16;
      hid[row * 1536 + col] = (_Float16)gelu_tanh(acc[nt][r] + b1[col]);
    }
  }
}

// ---------------------------------------------------------------------------
// MLP2: M=50176, N=384, K=1536, bias + residual -> f32 output. grid (3, 784).
// ---------------------------------------------------------------------------
__global__ __launch_bounds__(256) void mlp2_gemm_kernel(const _Float16* __restrict__ hid,
                                                        const _Float16* __restrict__ w2t,
                                                        const float* __restrict__ b2,
                                                        const float* __restrict__ x1,
                                                        float* __restrict__ out) {
  __shared__ __align__(16) _Float16 sA[2 * 64 * 32];
  __shared__ __align__(16) _Float16 sB[2 * 128 * 32];
  const int mb = blockIdx.y;
  const int nblk = blockIdx.x;
  v8f acc[4];
  gemm_tile_64x128(hid + (size_t)mb * 64 * 1536, 1536,
                   w2t + (size_t)nblk * 128 * 1536, 1536, 1536, sA, sB, acc);
  const int w = threadIdx.x >> 5, mr = w >> 1, nc = w & 1;
  const int l = threadIdx.x & 31, m16 = l & 15, hf = l >> 4;
#pragma unroll
  for (int nt = 0; nt < 4; ++nt) {
#pragma unroll
    for (int r = 0; r < 8; ++r) {
      size_t row = (size_t)mb * 64 + mr * 16 + hf * 8 + r;
      int col = nblk * 128 + nc * 64 + nt * 16 + m16;
      size_t off = row * 384 + col;
      out[off] = x1[off] + acc[nt][r] + b2[col];
    }
  }
}

// ---------------------------------------------------------------------------
// Host launcher
// ---------------------------------------------------------------------------
extern "C" void kernel_launch(void* const* d_in, const int* in_sizes, int n_in,
                              void* d_out, int out_size, void* d_ws, size_t ws_size,
                              hipStream_t stream) {
  (void)in_sizes; (void)n_in; (void)out_size; (void)ws_size;
  const float* x        = (const float*)d_in[0];
  const float* mask     = (const float*)d_in[1];
  const float* n1_sc    = (const float*)d_in[2];
  const float* n1_bi    = (const float*)d_in[3];
  const float* qkv_w    = (const float*)d_in[4];
  const float* qkv_b    = (const float*)d_in[5];
  const float* rel_tab  = (const float*)d_in[6];
  const float* proj_w   = (const float*)d_in[7];
  const float* proj_b   = (const float*)d_in[8];
  const float* n2_sc    = (const float*)d_in[9];
  const float* n2_bi    = (const float*)d_in[10];
  const float* mlp_w1   = (const float*)d_in[11];
  const float* mlp_b1   = (const float*)d_in[12];
  const float* mlp_w2   = (const float*)d_in[13];
  const float* mlp_b2   = (const float*)d_in[14];
  float* out = (float*)d_out;

  // workspace carve-up (256B aligned)
  char* ws = (char*)d_ws;
  size_t off = 0;
  auto take = [&](size_t bytes) -> void* {
    void* p = ws + off;
    off += (bytes + 255) & ~(size_t)255;
    return p;
  };
  float*    stats1  = (float*)take(16 * 2 * sizeof(float));
  float*    stats2  = (float*)take(16 * 2 * sizeof(float));
  _Float16* qkv_wt  = (_Float16*)take((size_t)1152 * 384 * 2);
  _Float16* proj_wt = (_Float16*)take((size_t)384 * 384 * 2);
  _Float16* w1t     = (_Float16*)take((size_t)1536 * 384 * 2);
  _Float16* w2t     = (_Float16*)take((size_t)384 * 1536 * 2);
  _Float16* xw      = (_Float16*)take((size_t)1024 * 64 * 384 * 2);  // aliased: attn_out
  _Float16* qbuf    = (_Float16*)take((size_t)1024 * 12 * 64 * 32 * 2);  // aliased: h2
  _Float16* kbuf    = (_Float16*)take((size_t)1024 * 12 * 64 * 32 * 2);
  _Float16* vtbuf   = (_Float16*)take((size_t)1024 * 12 * 32 * 64 * 2);
  float*    x1      = (float*)take((size_t)50176 * 384 * sizeof(float));
  _Float16* hid     = (_Float16*)take((size_t)50176 * 1536 * 2);
  _Float16* attn_o  = xw;    // xw fully consumed by QKV GEMM before attention writes
  _Float16* h2      = qbuf;  // q fully consumed by attention before LN2 writes

  // 0) weight convert + transpose to f16 [N][K]
  conv_transpose_kernel<<<(384 * 1152 + 255) / 256, 256, 0, stream>>>(qkv_w, qkv_wt, 384, 1152);
  conv_transpose_kernel<<<(384 * 384 + 255) / 256, 256, 0, stream>>>(proj_w, proj_wt, 384, 384);
  conv_transpose_kernel<<<(384 * 1536 + 255) / 256, 256, 0, stream>>>(mlp_w1, w1t, 384, 1536);
  conv_transpose_kernel<<<(1536 * 384 + 255) / 256, 256, 0, stream>>>(mlp_w2, w2t, 1536, 384);

  // 1) LN1 stats + apply/shift/partition
  ln_stats_kernel<<<16, 256, 0, stream>>>(x, stats1);
  ln1_partition_kernel<<<1024, 256, 0, stream>>>(x, stats1, n1_sc, n1_bi, xw);

  // 2) QKV GEMM (M=64/window, N=1152, K=384)
  qkv_gemm_kernel<<<dim3(9, 1024), 256, 0, stream>>>(xw, qkv_wt, qkv_b, qbuf, kbuf, vtbuf);

  // 3) fused attention per (window, head)
  attn_kernel<<<1024 * 12, 128, 0, stream>>>(qbuf, kbuf, vtbuf, rel_tab, mask, attn_o);

  // 4) proj GEMM + reverse + unshift + residual -> x1
  proj_gemm_kernel<<<dim3(3, 1024), 256, 0, stream>>>(attn_o, proj_wt, proj_b, x, x1);

  // 5) LN2 + MLP + residual -> out
  ln_stats_kernel<<<16, 256, 0, stream>>>(x1, stats2);
  ln2_apply_kernel<<<(50176 * 384) / 256, 256, 0, stream>>>(x1, stats2, n2_sc, n2_bi, h2);
  mlp1_gemm_kernel<<<dim3(12, 784), 256, 0, stream>>>(h2, w1t, mlp_b1, hid);
  mlp2_gemm_kernel<<<dim3(3, 784), 256, 0, stream>>>(hid, w2t, mlp_b2, x1, out);
}
